// VectorQuantizer_41154376630606
// MI455X (gfx1250) — compile-verified, matrix operand order checked by reading, not run on hardware
//
#include <hip/hip_runtime.h>
#include <math.h>

// ---------------- constants ----------------
#define D_EMB   256
#define K_EMB   1024
#define N_ROWS  32768           // 32*32*32
#define Q_ELEMS 8388608         // N_ROWS * D_EMB
#define IDX_OFF (Q_ELEMS + 3)   // quantized, commit, embed, ent, then indices
#define INV_T   100.0f          // 1 / ENT_TEMP

typedef __attribute__((ext_vector_type(16))) __bf16 v16bf;
typedef __attribute__((ext_vector_type(8)))  float  v8f;

// ---------------- helpers ----------------
__device__ inline unsigned short f2bf(float x) {
    unsigned int u = __float_as_uint(x);
    unsigned int r = u + 0x7FFFu + ((u >> 16) & 1u);   // round-to-nearest-even
    return (unsigned short)(r >> 16);
}
__device__ inline unsigned int pack_bf2(float lo, float hi) {
    return (unsigned int)f2bf(lo) | ((unsigned int)f2bf(hi) << 16);
}

// A-fragment (16x32 bf16) from an f32 row. Lane l holds row m=l&15; element i maps
// to k = i + 8h (i<8) / i + 8 + 8h (i>=8), h = lane>>4  -> two contiguous 8-elem chunks.
__device__ inline v16bf load_frag_f32(const float* __restrict__ rowp, int kbase, int h) {
    union { v16bf bf; unsigned int u[8]; } f;
    const float4* p0 = (const float4*)(rowp + kbase + 8 * h);
    const float4* p1 = (const float4*)(rowp + kbase + 16 + 8 * h);
    float4 a0 = p0[0], a1 = p0[1];
    float4 b0 = p1[0], b1 = p1[1];
    f.u[0] = pack_bf2(a0.x, a0.y); f.u[1] = pack_bf2(a0.z, a0.w);
    f.u[2] = pack_bf2(a1.x, a1.y); f.u[3] = pack_bf2(a1.z, a1.w);
    f.u[4] = pack_bf2(b0.x, b0.y); f.u[5] = pack_bf2(b0.z, b0.w);
    f.u[6] = pack_bf2(b1.x, b1.y); f.u[7] = pack_bf2(b1.z, b1.w);
    return f.bf;
}

// B-fragment (32x16 bf16) from pre-converted bf16 codebook row (lane l holds column
// n = l&15, same k mapping as A). Two 16-byte loads.
__device__ inline v16bf load_frag_bf16(const unsigned short* __restrict__ rowp, int kbase, int h) {
    union { v16bf bf; uint4 q[2]; } f;
    f.q[0] = *(const uint4*)(rowp + kbase + 8 * h);
    f.q[1] = *(const uint4*)(rowp + kbase + 16 + 8 * h);
    return f.bf;
}

// ---------------- kernel 0: codebook prep ----------------
__global__ void vq_prep(const float* __restrict__ cb,
                        unsigned short* __restrict__ cb16,
                        float* __restrict__ cnorm,
                        float* __restrict__ avgp,
                        float* __restrict__ scal) {
    const int k = blockIdx.x;          // code row
    const int t = threadIdx.x;         // d
    float c = cb[k * D_EMB + t];
    cb16[k * D_EMB + t] = f2bf(c);
    float s = c * c;
    #pragma unroll
    for (int off = 16; off; off >>= 1) s += __shfl_xor(s, off, 32);
    __shared__ float part[8];
    if ((t & 31) == 0) part[t >> 5] = s;
    __syncthreads();
    if (t == 0) {
        float tot = 0.f;
        #pragma unroll
        for (int i = 0; i < 8; ++i) tot += part[i];
        cnorm[k] = tot;
        avgp[k]  = 0.f;                 // zero prob accumulator each call
        if (k == 0) { scal[0] = 0.f; scal[1] = 0.f; }
    }
}

// ---------------- kernel 1: fused GEMM + argmin + softmax/entropy + gather ----------------
__global__ void __launch_bounds__(256)
vq_main(const float* __restrict__ z,
        const float* __restrict__ cb,
        const unsigned short* __restrict__ cb16,
        const float* __restrict__ cnorm,
        float* __restrict__ avgp,
        float* __restrict__ scal,      // [0]=sample-entropy sum, [1]=mse sum
        float* __restrict__ out) {
    extern __shared__ float scores[];  // 16 rows x 1024 cols, f32 (64 KB)
    __shared__ int rowIdxS[16];

    const int tid     = threadIdx.x;
    const int wave    = tid >> 5;
    const int lane    = tid & 31;
    const int half    = lane >> 4;
    const int l16     = lane & 15;
    const int rowbase = blockIdx.x * 16;

    // ---- GEMM: score = ||c||^2 - 2 z.c  (bf16 WMMA for the cross term) ----
    const float* zrow = z + (size_t)(rowbase + l16) * D_EMB;
    v16bf afrag[8];
    #pragma unroll
    for (int ks = 0; ks < 8; ++ks)
        afrag[ks] = load_frag_f32(zrow, ks * 32, half);

    #pragma unroll
    for (int j = 0; j < 8; ++j) {
        const int ncol = wave * 128 + j * 16 + l16;       // code index 0..1023
        const unsigned short* brow = cb16 + (size_t)ncol * D_EMB;
        v8f acc = {0.f, 0.f, 0.f, 0.f, 0.f, 0.f, 0.f, 0.f};
        #pragma unroll
        for (int ks = 0; ks < 8; ++ks) {
            v16bf bfrag = load_frag_bf16(brow, ks * 32, half);
            acc = __builtin_amdgcn_wmma_f32_16x16x32_bf16(
                      false, afrag[ks], false, bfrag, (short)0, acc, false, false);
        }
        const float cn = cnorm[ncol];
        #pragma unroll
        for (int r = 0; r < 8; ++r)
            scores[(r + 8 * half) * K_EMB + ncol] = cn - 2.0f * acc[r];
    }
    __syncthreads();

    // ---- per-row argmin + softmax(-score/T) + entropy; 16 threads per row ----
    {
        const int r    = tid >> 4;
        const int part = tid & 15;
        float* srow = scores + r * K_EMB;

        float bestV = 3.4e38f; int bestI = 0;
        for (int i = 0; i < 64; ++i) {
            int c = i * 16 + part;
            float s = srow[c];
            if (s < bestV) { bestV = s; bestI = c; }
        }
        #pragma unroll
        for (int off = 8; off; off >>= 1) {
            float ov = __shfl_xor(bestV, off, 16);
            int   oi = __shfl_xor(bestI, off, 16);
            if (ov < bestV || (ov == bestV && oi < bestI)) { bestV = ov; bestI = oi; }
        }

        float sumE = 0.f, sumXE = 0.f;
        for (int i = 0; i < 64; ++i) {
            int c = i * 16 + part;
            float d = INV_T * (bestV - srow[c]);   // x - xmax  (<= 0)
            float e = __expf(d);
            sumE  += e;
            sumXE += d * e;
            srow[c] = e;                           // stash exp in place
        }
        #pragma unroll
        for (int off = 8; off; off >>= 1) {
            sumE  += __shfl_xor(sumE,  off, 16);
            sumXE += __shfl_xor(sumXE, off, 16);
        }
        const float invS = 1.0f / sumE;
        for (int i = 0; i < 64; ++i) srow[i * 16 + part] *= invS;   // probs

        if (part == 0) {
            rowIdxS[r] = bestI;
            out[IDX_OFF + rowbase + r] = (float)bestI;              // encoding index
            float ent = logf(sumE) - sumXE * invS;                  // -sum p*log p
            atomicAdd(&scal[0], ent);
        }
    }
    __syncthreads();

    // ---- per-WG prob column sums -> global avg-prob accumulator ----
    #pragma unroll
    for (int i = 0; i < 4; ++i) {
        int c = tid + 256 * i;
        float s = 0.f;
        #pragma unroll
        for (int m = 0; m < 16; ++m) s += scores[m * K_EMB + c];
        atomicAdd(&avgp[c], s);
    }

    // ---- gather quantized + mse partial ----
    {
        const int r    = tid >> 4;
        const int part = tid & 15;
        const int idx  = rowIdxS[r];
        const float4* qrow = (const float4*)(cb + (size_t)idx * D_EMB) + part * 4;
        const float4* zr4  = (const float4*)(z  + (size_t)(rowbase + r) * D_EMB) + part * 4;
        float4* orow       = (float4*)(out + (size_t)(rowbase + r) * D_EMB) + part * 4;
        float acc = 0.f;
        #pragma unroll
        for (int q = 0; q < 4; ++q) {
            float4 qv = qrow[q], zv = zr4[q];
            orow[q] = qv;
            float dx = qv.x - zv.x, dy = qv.y - zv.y;
            float dz = qv.z - zv.z, dw = qv.w - zv.w;
            acc += dx * dx + dy * dy + dz * dz + dw * dw;
        }
        #pragma unroll
        for (int off = 8; off; off >>= 1) acc += __shfl_xor(acc, off, 16);
        if (part == 0) atomicAdd(&scal[1], acc);
    }
}

// ---------------- kernel 2: finalize scalars ----------------
__global__ void vq_finalize(const float* __restrict__ avgp,
                            const float* __restrict__ scal,
                            float* __restrict__ out) {
    __shared__ float red[256];
    const int t = threadIdx.x;
    float acc = 0.f;
    #pragma unroll
    for (int i = 0; i < 4; ++i) {
        float p = avgp[t + 256 * i] * (1.0f / (float)N_ROWS);
        acc += p * logf(p + 1e-5f);
    }
    red[t] = acc;
    __syncthreads();
    for (int s = 128; s; s >>= 1) {
        if (t < s) red[t] += red[t + s];
        __syncthreads();
    }
    if (t == 0) {
        float avg_entropy    = -red[0];
        float sample_entropy = scal[0] * (1.0f / (float)N_ROWS);
        float mse            = scal[1] * (1.0f / (float)Q_ELEMS);
        out[Q_ELEMS + 0] = 0.25f * mse;                               // commitment
        out[Q_ELEMS + 1] = mse;                                       // embedding
        out[Q_ELEMS + 2] = 0.1f * (sample_entropy - avg_entropy);     // ent loss
    }
}

// ---------------- launch ----------------
extern "C" void kernel_launch(void* const* d_in, const int* in_sizes, int n_in,
                              void* d_out, int out_size, void* d_ws, size_t ws_size,
                              hipStream_t stream) {
    const float* z  = (const float*)d_in[0];   // [32,32,32,256] f32
    const float* cb = (const float*)d_in[1];   // [1024,256] f32
    float* out = (float*)d_out;

    // workspace layout
    char* w = (char*)d_ws;
    unsigned short* cb16 = (unsigned short*)w;                 // 1024*256*2 = 524288 B
    float* cnorm = (float*)(w + 524288);                       // 4096 B
    float* avgp  = (float*)(w + 528384);                       // 4096 B
    float* scal  = (float*)(w + 532480);                       // 8 B: entSum, mseSum

    vq_prep<<<K_EMB, D_EMB, 0, stream>>>(cb, cb16, cnorm, avgp, scal);
    vq_main<<<N_ROWS / 16, 256, 16 * K_EMB * sizeof(float), stream>>>(
        z, cb, cb16, cnorm, avgp, scal, out);
    vq_finalize<<<1, 256, 0, stream>>>(avgp, scal, out);
}
